// NMT_32461362823931
// MI455X (gfx1250) — matrix operand order repaired and checked by reading, not compiled, not run on hardware
//
#include <hip/hip_runtime.h>

// ---------------------------------------------------------------------------
// NMT encoder/decoder for MI455X (gfx1250): bf16 WMMA GEMMs + f32 accum.
// S=64 B=32 L=64 H_ENC=512 H_DEC=1024 D_EMB=300(->320) VT=32000
// ---------------------------------------------------------------------------

typedef __bf16          bf16_t;
typedef __attribute__((ext_vector_type(16))) __bf16          v16bf;
typedef __attribute__((ext_vector_type(8)))  float           v8f;
typedef __attribute__((ext_vector_type(8)))  unsigned short  v8u16;
typedef __attribute__((ext_vector_type(16))) unsigned short  v16u16;

__device__ __forceinline__ unsigned short f2bf_u(float f) {
    unsigned int u = __float_as_uint(f);
    unsigned int r = (u + 0x7FFFu + ((u >> 16) & 1u)) >> 16;
    return (unsigned short)r;
}
__device__ __forceinline__ float sigm(float x) { return 1.f / (1.f + __expf(-x)); }

// ---------------------------------------------------------------------------
// Generic WMMA GEMM:  C[M x N] = act( A1[M x K1] @ W1[N x K1]^T
//                                   + A2[M x K2] @ W2[N x K2]^T
//                                   + bias[N] + Cin[M x ldcin] )
// A, W in bf16 (stored as u16), row-major, K multiple of 32.
// Block = 128 threads = 4 waves; each wave owns a 16x16 tile of a 32x32 block.
// ---------------------------------------------------------------------------
__global__ __launch_bounds__(128)
void k_wmma_gemm(const unsigned short* __restrict__ A1, int lda1, int K1,
                 const unsigned short* __restrict__ W1,
                 const unsigned short* __restrict__ A2, int lda2, int K2,
                 const unsigned short* __restrict__ W2,
                 const float* __restrict__ bias,
                 const float* __restrict__ Cin, int ldcin,
                 float* __restrict__ Cout, int ldcout,
                 unsigned short* __restrict__ Cbf, int ldcbf,
                 int M, int act)
{
    const int tid  = threadIdx.x;
    const int lane = tid & 31;
    const int wave = tid >> 5;
    const int mbase = blockIdx.y * 32 + (wave & 1) * 16;
    const int nbase = blockIdx.x * 32 + (wave >> 1) * 16;
    if (mbase >= M) return;                       // uniform per wave

    const int arow = mbase + (lane & 15);
    const int wrow = nbase + (lane & 15);
    const int akoff = (lane & 16) ? 16 : 0;       // A: lanes 0-15 K{0-15},     16-31 K{16-31}
    const int bkoff = (lane & 16) ? 16 : 0;       // B: lanes 0-15 K{0-15},     16-31 K{16-31}

    v8f acc = {};

    {
        const unsigned short* ap = A1 + (size_t)arow * lda1 + akoff;
        const unsigned short* wp = W1 + (size_t)wrow * K1  + bkoff;
        for (int k = 0; k < K1; k += 32) {
            __builtin_prefetch(wp + k + 64, 0, 0);
            v8u16 lo = *(const v8u16*)(ap + k);
            v8u16 hi = *(const v8u16*)(ap + k + 8);
            v16u16 au;
#pragma unroll
            for (int i = 0; i < 8; ++i) { au[i] = lo[i]; au[i + 8] = hi[i]; }
            v16bf af = __builtin_bit_cast(v16bf, au);
            v16bf bfm = __builtin_bit_cast(v16bf, *(const v16u16*)(wp + k));
            acc = __builtin_amdgcn_wmma_f32_16x16x32_bf16(
                      false, af, false, bfm, (short)0, acc, false, false);
        }
    }
    if (A2) {
        const unsigned short* ap = A2 + (size_t)arow * lda2 + akoff;
        const unsigned short* wp = W2 + (size_t)wrow * K2  + bkoff;
        for (int k = 0; k < K2; k += 32) {
            __builtin_prefetch(wp + k + 64, 0, 0);
            v8u16 lo = *(const v8u16*)(ap + k);
            v8u16 hi = *(const v8u16*)(ap + k + 8);
            v16u16 au;
#pragma unroll
            for (int i = 0; i < 8; ++i) { au[i] = lo[i]; au[i + 8] = hi[i]; }
            v16bf af = __builtin_bit_cast(v16bf, au);
            v16bf bfm = __builtin_bit_cast(v16bf, *(const v16u16*)(wp + k));
            acc = __builtin_amdgcn_wmma_f32_16x16x32_bf16(
                      false, af, false, bfm, (short)0, acc, false, false);
        }
    }

    // C/D layout: lanes 0-15 hold N=lane, rows M=r; lanes 16-31 hold N=lane-16, rows M=8+r
    const int col   = nbase + (lane & 15);
    const int rbase = mbase + ((lane >> 4) * 8);
    const float bb  = bias ? bias[col] : 0.f;
#pragma unroll
    for (int r = 0; r < 8; ++r) {
        int row = rbase + r;
        if (row < M) {
            float v = acc[r] + bb;
            if (Cin)  v += Cin[(size_t)row * ldcin + col];
            if (act == 1) v = tanhf(v);
            if (Cout) Cout[(size_t)row * ldcout + col] = v;
            if (Cbf)  Cbf [(size_t)row * ldcbf  + col] = f2bf_u(v);
        }
    }
}

// ---------------------------------------------------------------------------
// Elementwise / reduction kernels
// ---------------------------------------------------------------------------
__global__ void k_lstm_act(const float* __restrict__ gates, int H,
                           float* __restrict__ c,
                           float* __restrict__ h_f32, int h_stride, int h_off,
                           unsigned short* __restrict__ h_bf, int hb_stride)
{
    int idx = blockIdx.x * blockDim.x + threadIdx.x;
    if (idx >= 32 * H) return;
    int b = idx / H, j = idx % H;
    const float* g = gates + (size_t)b * 4 * H;
    float gi = g[j], gf = g[H + j], gg = g[2 * H + j], go = g[3 * H + j];
    float cv = sigm(gf) * c[idx] + sigm(gi) * tanhf(gg);
    float hv = sigm(go) * tanhf(cv);
    c[idx] = cv;
    if (h_f32) h_f32[(size_t)b * h_stride + h_off + j] = hv;
    h_bf[(size_t)b * hb_stride + j] = f2bf_u(hv);
}

__global__ __launch_bounds__(256)
void k_attention(const float* __restrict__ enc_out,   // [64][32][1024]
                 const float* __restrict__ semi,      // [32][1024]
                 const unsigned short* __restrict__ h_bf, // [32][1024]
                 unsigned short* __restrict__ xcat)   // [32][2048] = [s_tilde | h]
{
    __shared__ float part[256];
    __shared__ float sc[64];
    __shared__ float red[2];
    const int b = blockIdx.x, tid = threadIdx.x;
    const int s = tid >> 2, q = tid & 3;
    const float* er = enc_out + ((size_t)s * 32 + b) * 1024;
    const float* sm = semi + (size_t)b * 1024;
    float acc = 0.f;
    for (int d = q * 256; d < q * 256 + 256; ++d) acc += er[d] * sm[d];
    part[tid] = acc;
    __syncthreads();
    if (q == 0) sc[s] = part[tid] + part[tid + 1] + part[tid + 2] + part[tid + 3];
    __syncthreads();
    if (tid == 0) { float mx = sc[0]; for (int i = 1; i < 64; ++i) mx = fmaxf(mx, sc[i]); red[0] = mx; }
    __syncthreads();
    if (tid < 64) sc[tid] = __expf(sc[tid] - red[0]);
    __syncthreads();
    if (tid == 0) { float s2 = 0.f; for (int i = 0; i < 64; ++i) s2 += sc[i]; red[1] = 1.f / s2; }
    __syncthreads();
    for (int d = tid; d < 1024; d += 256) {
        float a2 = 0.f;
        for (int s2 = 0; s2 < 64; ++s2)
            a2 += sc[s2] * enc_out[((size_t)s2 * 32 + b) * 1024 + d];
        xcat[(size_t)b * 2048 + d] = f2bf_u(a2 * red[1]);
    }
    for (int d = tid; d < 1024; d += 256)
        xcat[(size_t)b * 2048 + 1024 + d] = h_bf[(size_t)b * 1024 + d];
}

__global__ __launch_bounds__(512)
void k_logsoftmax(const float* __restrict__ logits, float* __restrict__ out)
{
    __shared__ float red[512];
    const int b = blockIdx.x, tid = threadIdx.x;
    const float* row = logits + (size_t)b * 32000;
    float mx = -3.4e38f;
    for (int n = tid; n < 32000; n += 512) mx = fmaxf(mx, row[n]);
    red[tid] = mx; __syncthreads();
    for (int o = 256; o > 0; o >>= 1) { if (tid < o) red[tid] = fmaxf(red[tid], red[tid + o]); __syncthreads(); }
    mx = red[0]; __syncthreads();
    float sm = 0.f;
    for (int n = tid; n < 32000; n += 512) sm += __expf(row[n] - mx);
    red[tid] = sm; __syncthreads();
    for (int o = 256; o > 0; o >>= 1) { if (tid < o) red[tid] += red[tid + o]; __syncthreads(); }
    float lse = mx + __logf(red[0]);
    for (int n = tid; n < 32000; n += 512) out[(size_t)b * 32000 + n] = row[n] - lse;
}

__global__ void k_build_hc(const float* __restrict__ enc_out,
                           const float* __restrict__ c_f, const float* __restrict__ c_b,
                           unsigned short* __restrict__ h_bf_d, float* __restrict__ c_d)
{
    int idx = blockIdx.x * blockDim.x + threadIdx.x;   // 32*512
    if (idx >= 32 * 512) return;
    int b = idx / 512, j = idx % 512;
    float hf = enc_out[((size_t)63 * 32 + b) * 1024 + j];
    float hb = enc_out[((size_t)b) * 1024 + 512 + j];  // out_b[0]
    h_bf_d[(size_t)b * 1024 + 2 * j]     = f2bf_u(hf);
    h_bf_d[(size_t)b * 1024 + 2 * j + 1] = f2bf_u(hb);
    c_d[(size_t)b * 1024 + 2 * j]     = c_f[(size_t)b * 512 + j];
    c_d[(size_t)b * 1024 + 2 * j + 1] = c_b[(size_t)b * 512 + j];
}

__global__ void k_convert_pad(const float* __restrict__ src, unsigned short* __restrict__ dst,
                              int N, int K, int Kp)
{
    long long idx = (long long)blockIdx.x * blockDim.x + threadIdx.x;
    if (idx >= (long long)N * Kp) return;
    int n = (int)(idx / Kp), k = (int)(idx % Kp);
    dst[idx] = (k < K) ? f2bf_u(src[(size_t)n * K + k]) : (unsigned short)0;
}

__global__ void k_gather_src(const float* __restrict__ eemb, const int* __restrict__ srcidx,
                             unsigned short* __restrict__ dst)
{
    int idx = blockIdx.x * blockDim.x + threadIdx.x;   // 2048*320
    if (idx >= 2048 * 320) return;
    int sb = idx / 320, d = idx % 320;
    int tok = srcidx[sb];
    dst[idx] = (d < 300) ? f2bf_u(eemb[(size_t)tok * 300 + d]) : (unsigned short)0;
}

__global__ void k_fill_xemb(const float* __restrict__ demb, const int* __restrict__ trg,
                            unsigned short* __restrict__ xbuf)
{
    int idx = blockIdx.x * blockDim.x + threadIdx.x;   // 63*32*320
    if (idx >= 63 * 32 * 320) return;
    int d = idx % 320, tb = idx / 320;
    int tok = trg[tb];
    xbuf[(size_t)tb * 1344 + 1024 + d] = (d < 300) ? f2bf_u(demb[(size_t)tok * 300 + d])
                                                   : (unsigned short)0;
}

__global__ void k_zero_f32(float* p, long long n)
{ long long i = (long long)blockIdx.x * blockDim.x + threadIdx.x; if (i < n) p[i] = 0.f; }
__global__ void k_zero_u16(unsigned short* p, long long n)
{ long long i = (long long)blockIdx.x * blockDim.x + threadIdx.x; if (i < n) p[i] = 0; }

// ---------------------------------------------------------------------------
// Host orchestration
// ---------------------------------------------------------------------------
extern "C" void kernel_launch(void* const* d_in, const int* in_sizes, int n_in,
                              void* d_out, int out_size, void* d_ws, size_t ws_size,
                              hipStream_t stream)
{
    (void)in_sizes; (void)n_in; (void)out_size; (void)ws_size;
    const float* eemb = (const float*)d_in[0];
    const float* wihf = (const float*)d_in[1];
    const float* whhf = (const float*)d_in[2];
    const float* ebf  = (const float*)d_in[3];
    const float* wihb = (const float*)d_in[4];
    const float* whhb = (const float*)d_in[5];
    const float* ebb  = (const float*)d_in[6];
    const float* awi  = (const float*)d_in[7];
    const float* awo  = (const float*)d_in[8];
    const float* dwih = (const float*)d_in[9];
    const float* dwhh = (const float*)d_in[10];
    const float* db   = (const float*)d_in[11];
    const float* demb = (const float*)d_in[12];
    const float* genw = (const float*)d_in[13];
    const float* genb = (const float*)d_in[14];
    const int*   src  = (const int*)d_in[15];
    const int*   trg  = (const int*)d_in[16];
    float* out = (float*)d_out;

    // ---- workspace carving (256B aligned) ----
    size_t off = 0; char* base = (char*)d_ws;
    auto carve = [&](size_t bytes) -> void* {
        off = (off + 255) & ~(size_t)255; void* p = base + off; off += bytes; return p;
    };
    unsigned short* wihf_bf = (unsigned short*)carve((size_t)2048 * 320 * 2);
    unsigned short* wihb_bf = (unsigned short*)carve((size_t)2048 * 320 * 2);
    unsigned short* whhf_bf = (unsigned short*)carve((size_t)2048 * 512 * 2);
    unsigned short* whhb_bf = (unsigned short*)carve((size_t)2048 * 512 * 2);
    unsigned short* awi_bf  = (unsigned short*)carve((size_t)1024 * 1024 * 2);
    unsigned short* awo_bf  = (unsigned short*)carve((size_t)1024 * 2048 * 2);
    unsigned short* dwih_bf = (unsigned short*)carve((size_t)4096 * 1344 * 2);
    unsigned short* dwhh_bf = (unsigned short*)carve((size_t)4096 * 1024 * 2);
    unsigned short* genw_bf = (unsigned short*)carve((size_t)32000 * 1024 * 2);
    unsigned short* semb_bf = (unsigned short*)carve((size_t)2048 * 320 * 2);
    float* xWf     = (float*)carve((size_t)2048 * 2048 * 4);
    float* xWb     = (float*)carve((size_t)2048 * 2048 * 4);
    float* enc_out = (float*)carve((size_t)64 * 32 * 1024 * 4);
    float* gates   = (float*)carve((size_t)32 * 4096 * 4);
    unsigned short* hbf_f = (unsigned short*)carve((size_t)32 * 512 * 2);
    unsigned short* hbf_b = (unsigned short*)carve((size_t)32 * 512 * 2);
    unsigned short* hbf_d = (unsigned short*)carve((size_t)32 * 1024 * 2);
    float* c_f  = (float*)carve((size_t)32 * 512 * 4);
    float* c_b  = (float*)carve((size_t)32 * 512 * 4);
    float* c_d  = (float*)carve((size_t)32 * 1024 * 4);
    float* semi = (float*)carve((size_t)32 * 1024 * 4);
    unsigned short* xcat = (unsigned short*)carve((size_t)32 * 2048 * 2);
    unsigned short* xbuf = (unsigned short*)carve((size_t)63 * 32 * 1344 * 2);
    float* logits = (float*)carve((size_t)32 * 32000 * 4);

    auto gemm = [&](const void* A1, int lda1, int K1, const void* W1,
                    const void* A2, int lda2, int K2, const void* W2,
                    const float* bias, const float* Cin, int ldcin,
                    float* Cout, int ldcout, void* Cbf, int ldcbf,
                    int M, int N, int act) {
        dim3 g(N / 32, (M + 31) / 32);
        k_wmma_gemm<<<g, 128, 0, stream>>>(
            (const unsigned short*)A1, lda1, K1, (const unsigned short*)W1,
            (const unsigned short*)A2, lda2, K2, (const unsigned short*)W2,
            bias, Cin, ldcin, Cout, ldcout, (unsigned short*)Cbf, ldcbf, M, act);
    };
    auto conv = [&](const float* s, unsigned short* d, int N, int K, int Kp) {
        long long t = (long long)N * Kp;
        k_convert_pad<<<(unsigned)((t + 255) / 256), 256, 0, stream>>>(s, d, N, K, Kp);
    };
    auto zf = [&](float* p, long long n) { k_zero_f32<<<(unsigned)((n + 255) / 256), 256, 0, stream>>>(p, n); };
    auto zu = [&](unsigned short* p, long long n) { k_zero_u16<<<(unsigned)((n + 255) / 256), 256, 0, stream>>>(p, n); };

    // ---- weight conversion fp32 -> bf16 (K padded to /32) ----
    conv(wihf, wihf_bf, 2048, 300, 320);
    conv(wihb, wihb_bf, 2048, 300, 320);
    conv(whhf, whhf_bf, 2048, 512, 512);
    conv(whhb, whhb_bf, 2048, 512, 512);
    conv(awi,  awi_bf,  1024, 1024, 1024);
    conv(awo,  awo_bf,  1024, 2048, 2048);
    conv(dwih, dwih_bf, 4096, 1324, 1344);
    conv(dwhh, dwhh_bf, 4096, 1024, 1024);
    conv(genw, genw_bf, 32000, 1024, 1024);

    // ---- embeddings ----
    k_gather_src<<<(2048 * 320 + 255) / 256, 256, 0, stream>>>(eemb, src, semb_bf);
    k_fill_xemb<<<(63 * 32 * 320 + 255) / 256, 256, 0, stream>>>(demb, trg, xbuf);

    // ---- encoder input projections for ALL timesteps (big parallel GEMMs) ----
    gemm(semb_bf, 320, 320, wihf_bf, nullptr, 0, 0, nullptr,
         ebf, nullptr, 0, xWf, 2048, nullptr, 0, 2048, 2048, 0);
    gemm(semb_bf, 320, 320, wihb_bf, nullptr, 0, 0, nullptr,
         ebb, nullptr, 0, xWb, 2048, nullptr, 0, 2048, 2048, 0);

    // ---- encoder recurrences ----
    zu(hbf_f, 32 * 512); zf(c_f, 32 * 512);
    zu(hbf_b, 32 * 512); zf(c_b, 32 * 512);
    for (int s = 0; s < 64; ++s) {
        gemm(hbf_f, 512, 512, whhf_bf, nullptr, 0, 0, nullptr,
             nullptr, xWf + (size_t)s * 32 * 2048, 2048, gates, 2048, nullptr, 0, 32, 2048, 0);
        k_lstm_act<<<(32 * 512 + 255) / 256, 256, 0, stream>>>(
            gates, 512, c_f, enc_out + (size_t)s * 32 * 1024, 1024, 0, hbf_f, 512);
    }
    for (int s = 63; s >= 0; --s) {
        gemm(hbf_b, 512, 512, whhb_bf, nullptr, 0, 0, nullptr,
             nullptr, xWb + (size_t)s * 32 * 2048, 2048, gates, 2048, nullptr, 0, 32, 2048, 0);
        k_lstm_act<<<(32 * 512 + 255) / 256, 256, 0, stream>>>(
            gates, 512, c_b, enc_out + (size_t)s * 32 * 1024, 1024, 512, hbf_b, 512);
    }

    // ---- initial decoder state (interleave fwd/bwd) ----
    k_build_hc<<<(32 * 512 + 255) / 256, 256, 0, stream>>>(enc_out, c_f, c_b, hbf_d, c_d);

    // ---- output row 0 = zeros ----
    zf(out, (long long)32 * 32000);

    // ---- decoder steps ----
    for (int t = 0; t < 63; ++t) {
        unsigned short* xt = xbuf + (size_t)t * 32 * 1344;
        // semi = h @ attn_wi^T
        gemm(hbf_d, 1024, 1024, awi_bf, nullptr, 0, 0, nullptr,
             nullptr, nullptr, 0, semi, 1024, nullptr, 0, 32, 1024, 0);
        // attention: scores softmax over S, s_tilde; xcat = [s_tilde | h] (bf16)
        k_attention<<<32, 256, 0, stream>>>(enc_out, semi, hbf_d, xcat);
        // c_t = tanh(xcat @ attn_wo^T) -> bf16 into xbuf[t][:, 0:1024]
        gemm(xcat, 2048, 2048, awo_bf, nullptr, 0, 0, nullptr,
             nullptr, nullptr, 0, nullptr, 0, xt, 1344, 32, 1024, 1);
        // gates = x @ dec_w_ih^T + h @ dec_w_hh^T + b
        gemm(xt, 1344, 1344, dwih_bf, hbf_d, 1024, 1024, dwhh_bf,
             db, nullptr, 0, gates, 4096, nullptr, 0, 32, 4096, 0);
        k_lstm_act<<<(32 * 1024 + 255) / 256, 256, 0, stream>>>(
            gates, 1024, c_d, nullptr, 0, 0, hbf_d, 1024);
        // logits = h @ gen_w^T + gen_b   (gen_w bf16 = 65 MB -> L2 resident)
        gemm(hbf_d, 1024, 1024, genw_bf, nullptr, 0, 0, nullptr,
             genb, nullptr, 0, logits, 32000, nullptr, 0, 32, 32000, 0);
        // out[t+1] = log_softmax(logits)
        k_logsoftmax<<<32, 512, 0, stream>>>(logits, out + (size_t)(t + 1) * 32 * 32000);
    }
}
